// GraphDecoder_34900904247654
// MI455X (gfx1250) — matrix operand order repaired and checked by reading, not compiled, not run on hardware
//
#include <hip/hip_runtime.h>
#include <cstdint>
#include <cstddef>

typedef __attribute__((ext_vector_type(16))) _Float16 v16h;
typedef __attribute__((ext_vector_type(8)))  _Float16 v8h;
typedef __attribute__((ext_vector_type(8)))  float    v8f;

#define LATENT    256
#define HIDDEN    512
#define NODE_F    128
#define MAX_NODES 50
#define BATCH     128
#define NOUT      (MAX_NODES*NODE_F)   /* 6400 */
#define NEDGE     1225
#define EHID      (HIDDEN/2)           /* 256 */

#define EDGE_BLK  64                   /* edges per block (4 waves x 16) */

__device__ __forceinline__ float sigmoidf(float x){ return 1.0f/(1.0f+__expf(-x)); }

/* ---------------- fragment loaders (wave32; contiguous k order shared by A and B) ----------------
 * A 16x32 f16, row-major source (lda in elements), 16B-aligned chunks:
 *   lane L: M = L&15 ; element e -> K = k0 + e + 16*(L>>4)
 * B 32x16 f16 from transposed weights Bt (N rows x K cols, row-major):
 *   lane L: N = n0 + (L&15) ; element e -> K = k0 + e + 16*(L>>4)  (32B contig)
 * C/D f32 16x16: lane L col = L&15 ; row r-th VGPR = r + 8*(L>>4)
 * --------------------------------------------------------------------------- */
__device__ __forceinline__ v16h load_a_frag(const _Float16* __restrict__ A, int lda,
                                            int m0, int k0, int lane){
  int m  = m0 + (lane & 15);
  int hk = (lane >> 4) * 16;
  const _Float16* p = A + (size_t)m * lda + k0 + hk;
  v8h lo = *(const v8h*)(p);
  v8h hi = *(const v8h*)(p + 8);
  v16h r;
#pragma unroll
  for (int e = 0; e < 8; ++e){ r[e] = lo[e]; r[e+8] = hi[e]; }
  return r;
}

__device__ __forceinline__ v16h load_b_frag(const _Float16* __restrict__ Bt, int ldb,
                                            int n0, int k0, int lane){
  int n = n0 + (lane & 15);
  const _Float16* p = Bt + (size_t)n * ldb + k0 + (lane >> 4) * 16;
  return *(const v16h*)(p);   /* 32B aligned: ldb mult of 128, k0 mult of 32 */
}

/* -------------------------- small utility kernels -------------------------- */
__global__ void k_cvt(const float* __restrict__ s, _Float16* __restrict__ d, int n){
  int i = blockIdx.x*256 + threadIdx.x;
  if (i < n) d[i] = (_Float16)s[i];
}

/* f32 (K x N) row-major  ->  f16 (N x K) row-major  (weight transpose) */
__global__ void k_cvt_t(const float* __restrict__ s, _Float16* __restrict__ d, int K, int N){
  int i = blockIdx.x*256 + threadIdx.x;
  if (i < K*N){ int n = i / K, k = i % K; d[i] = (_Float16)s[(size_t)k*N + n]; }
}

__global__ void k_edge_idx(int* __restrict__ ii, int* __restrict__ jj){
  int e = blockIdx.x*256 + threadIdx.x;
  if (e < NEDGE){
    int i = 0, rem = e;
    while (rem >= (MAX_NODES - 1 - i)) { rem -= (MAX_NODES - 1 - i); ++i; }
    ii[e] = i; jj[e] = i + 1 + rem;
  }
}

/* ------------------------------ WMMA GEMM ----------------------------------
 * C(M,N) = act(A(M,K)_f16 @ B(K,N) + bias),  B given transposed (N x K) f16.
 * One wave (32 thr) computes a 16x64 tile: 1 A-frag reused by 4 B-frags.
 * M % 16 == 0, N % 64 == 0, K % 32 == 0 (true for every call we make).
 * act: 0=none 1=relu 2=sigmoid ; Cf (f32) and/or Ch (f16) outputs.
 * --------------------------------------------------------------------------- */
__global__ __launch_bounds__(32) void k_gemm_wmma(
    const _Float16* __restrict__ A, int lda,
    const _Float16* __restrict__ Bt, int ldb,
    const float* __restrict__ bias,
    float* __restrict__ Cf, int actF,
    _Float16* __restrict__ Ch, int actH,
    int M, int N, int K)
{
  int lane = threadIdx.x;
  int m0 = blockIdx.y * 16;
  int n0 = blockIdx.x * 64;
  v8f acc[4] = {};
  for (int k0 = 0; k0 < K; k0 += 32){
    v16h a = load_a_frag(A, lda, m0, k0, lane);
#pragma unroll
    for (int t = 0; t < 4; ++t){
      v16h b = load_b_frag(Bt, ldb, n0 + 16*t, k0, lane);
      acc[t] = __builtin_amdgcn_wmma_f32_16x16x32_f16(false, a, false, b,
                                                      (short)0, acc[t], false, false);
    }
  }
#pragma unroll
  for (int t = 0; t < 4; ++t){
    int n = n0 + 16*t + (lane & 15);
    float bv = bias ? bias[n] : 0.0f;
#pragma unroll
    for (int r = 0; r < 8; ++r){
      int m = m0 + r + 8*(lane >> 4);
      float v = acc[t][r] + bv;
      if (Cf){
        float o = v;
        if (actF == 1) o = fmaxf(o, 0.0f);
        else if (actF == 2) o = sigmoidf(o);
        Cf[(size_t)m*N + n] = o;
      }
      if (Ch){
        float o = (actH == 1) ? fmaxf(v, 0.0f) : v;
        Ch[(size_t)m*N + n] = (_Float16)o;
      }
    }
  }
}

/* --------------------------- fused edge kernel -----------------------------
 * grid = (BATCH, ceil(NEDGE/64)); 128 threads = 4 waves per block.
 * phase1: e1[64 x 512] = relu(Pz[b] + Pi[b,i] + Pj[b,j]) -> LDS f16 (64 KB)
 * phase2: per wave, its 16-edge A-tile vs ew2t; all 4 waves sweep the same
 *         (nt,kt) B-fragment sequence -> B hits WGP$ instead of L2.
 *         Layer-3 dot (h2 . ew3) folded into the WMMA epilogue: each lane
 *         accumulates relu(acc+eb2[n])*ew3[n] into 8 row-partials, then a
 *         shfl_xor tree (1/2/4/8) reduces over the 16 columns per lane-half.
 * --------------------------------------------------------------------------- */
__global__ __launch_bounds__(128) void k_edge(
    const float* __restrict__ Pz, const float* __restrict__ Pi,
    const float* __restrict__ Pj,
    const _Float16* __restrict__ ew2t,           /* 256 x 512, f16 transposed */
    const float* __restrict__ eb2, const float* __restrict__ ew3,
    const float* __restrict__ eb3,
    const int* __restrict__ ii, const int* __restrict__ jj,
    float* __restrict__ outE)
{
  __shared__ _Float16 e1[EDGE_BLK * HIDDEN];     /* 64 KB */

  int b    = blockIdx.x;
  int e0   = blockIdx.y * EDGE_BLK;
  int tid  = threadIdx.x;
  int lane = tid & 31;
  int wave = tid >> 5;

  /* phase 1: cooperatively build 64 activated edge rows */
  const float* pz = Pz + (size_t)b * HIDDEN;
  for (int m = 0; m < EDGE_BLK; ++m){
    int e = e0 + m; if (e > NEDGE - 1) e = NEDGE - 1;   /* clamp reads */
    const float* pi = Pi + ((size_t)b * MAX_NODES + ii[e]) * HIDDEN;
    const float* pj = Pj + ((size_t)b * MAX_NODES + jj[e]) * HIDDEN;
    for (int k = tid; k < HIDDEN; k += 128){
      float v = pz[k] + pi[k] + pj[k];                  /* eb1 folded in Pz */
      e1[m * HIDDEN + k] = (_Float16)fmaxf(v, 0.0f);
    }
  }
  __syncthreads();

  /* phase 2: (16 x 512) @ (512 x 256) per wave + fused 256->1 dot */
  int m0 = wave * 16;                  /* this wave's row offset in e1 */
  float pr[8] = {0,0,0,0,0,0,0,0};     /* per-row partial dot with ew3 */

  for (int nt = 0; nt < EHID/16; ++nt){
    v8f acc = {};
#pragma unroll 4
    for (int kt = 0; kt < HIDDEN/32; ++kt){
      v16h a  = load_a_frag(e1, HIDDEN, m0, kt*32, lane);      /* DS loads */
      v16h bf = load_b_frag(ew2t, HIDDEN, nt*16, kt*32, lane);
      acc = __builtin_amdgcn_wmma_f32_16x16x32_f16(false, a, false, bf,
                                                   (short)0, acc, false, false);
    }
    int n = nt*16 + (lane & 15);
    float bv = eb2[n];
    float wv = ew3[n];
#pragma unroll
    for (int r = 0; r < 8; ++r)
      pr[r] += fmaxf(acc[r] + bv, 0.0f) * wv;
  }

  /* reduce over the 16 columns held by each lane-half */
#pragma unroll
  for (int r = 0; r < 8; ++r){
    float v = pr[r];
    v += __shfl_xor(v, 1, 32);
    v += __shfl_xor(v, 2, 32);
    v += __shfl_xor(v, 4, 32);
    v += __shfl_xor(v, 8, 32);
    pr[r] = v;
  }

  if ((lane & 15) == 0){
    float b3 = eb3[0];
#pragma unroll
    for (int r = 0; r < 8; ++r){
      int m = r + 8*(lane >> 4);
      int e = e0 + m0 + m;
      if (e < NEDGE)
        outE[(size_t)b * NEDGE + e] = sigmoidf(pr[r] + b3);
    }
  }
}

/* ------------------------------- launcher ---------------------------------- */
extern "C" void kernel_launch(void* const* d_in, const int* in_sizes, int n_in,
                              void* d_out, int out_size, void* d_ws, size_t ws_size,
                              hipStream_t stream)
{
  (void)in_sizes; (void)n_in; (void)out_size; (void)ws_size;
  const float* z   = (const float*)d_in[0];
  const float* nw1 = (const float*)d_in[1];  const float* nb1 = (const float*)d_in[2];
  const float* nw2 = (const float*)d_in[3];  const float* nb2 = (const float*)d_in[4];
  const float* nw3 = (const float*)d_in[5];  const float* nb3 = (const float*)d_in[6];
  const float* ew1 = (const float*)d_in[7];  const float* eb1 = (const float*)d_in[8];
  const float* ew2 = (const float*)d_in[9];  const float* eb2 = (const float*)d_in[10];
  const float* ew3 = (const float*)d_in[11]; const float* eb3 = (const float*)d_in[12];

  float* out_nodes = (float*)d_out;                              /* 128*6400 */
  float* out_edges = (float*)d_out + (size_t)BATCH * NOUT;       /* 128*1225 */

  /* workspace carve (256B aligned), ~37 MB total */
  char* cur = (char*)d_ws;
  auto carve = [&](size_t bytes) -> void* {
    void* p = (void*)cur;
    cur += (bytes + 255) & ~(size_t)255;
    return p;
  };
  _Float16* zh    = (_Float16*)carve((size_t)BATCH*LATENT*2);
  _Float16* nw1t  = (_Float16*)carve((size_t)HIDDEN*LATENT*2);
  _Float16* nw2t  = (_Float16*)carve((size_t)HIDDEN*HIDDEN*2);
  _Float16* nw3t  = (_Float16*)carve((size_t)NOUT*HIDDEN*2);
  _Float16* ew1t  = (_Float16*)carve((size_t)HIDDEN*HIDDEN*2);
  _Float16* ew2t  = (_Float16*)carve((size_t)EHID*HIDDEN*2);
  _Float16* h1h   = (_Float16*)carve((size_t)BATCH*HIDDEN*2);
  _Float16* h2h   = (_Float16*)carve((size_t)BATCH*HIDDEN*2);
  _Float16* nfh   = (_Float16*)carve((size_t)BATCH*NOUT*2);
  float*    Pz    = (float*)carve((size_t)BATCH*HIDDEN*4);
  float*    Pi    = (float*)carve((size_t)BATCH*MAX_NODES*HIDDEN*4);
  float*    Pj    = (float*)carve((size_t)BATCH*MAX_NODES*HIDDEN*4);
  int*      iiW   = (int*)carve((size_t)NEDGE*4);
  int*      jjW   = (int*)carve((size_t)NEDGE*4);

  /* 1. conversions */
  { int n = BATCH*LATENT;  k_cvt  <<<(n+255)/256, 256, 0, stream>>>(z, zh, n); }
  { int n = LATENT*HIDDEN; k_cvt_t<<<(n+255)/256, 256, 0, stream>>>(nw1, nw1t, LATENT, HIDDEN); }
  { int n = HIDDEN*HIDDEN; k_cvt_t<<<(n+255)/256, 256, 0, stream>>>(nw2, nw2t, HIDDEN, HIDDEN); }
  { int n = HIDDEN*NOUT;   k_cvt_t<<<(n+255)/256, 256, 0, stream>>>(nw3, nw3t, HIDDEN, NOUT); }
  { int n = HIDDEN*HIDDEN; k_cvt_t<<<(n+255)/256, 256, 0, stream>>>(ew1, ew1t, HIDDEN, HIDDEN); }
  { int n = HIDDEN*EHID;   k_cvt_t<<<(n+255)/256, 256, 0, stream>>>(ew2, ew2t, HIDDEN, EHID); }
  k_edge_idx<<<(NEDGE+255)/256, 256, 0, stream>>>(iiW, jjW);

  /* 2. node MLP */
  k_gemm_wmma<<<dim3(HIDDEN/64, BATCH/16), 32, 0, stream>>>(
      zh, LATENT, nw1t, LATENT, nb1, nullptr, 0, h1h, 1, BATCH, HIDDEN, LATENT);
  k_gemm_wmma<<<dim3(HIDDEN/64, BATCH/16), 32, 0, stream>>>(
      h1h, HIDDEN, nw2t, HIDDEN, nb2, nullptr, 0, h2h, 1, BATCH, HIDDEN, HIDDEN);
  /* node logits: sigmoid -> out (f32), raw -> nfh (f16) for edge branch */
  k_gemm_wmma<<<dim3(NOUT/64, BATCH/16), 32, 0, stream>>>(
      h2h, HIDDEN, nw3t, HIDDEN, nb3, out_nodes, 2, nfh, 0, BATCH, NOUT, HIDDEN);

  /* 3. edge layer-1 decomposition: Pz (+eb1), Pi, Pj */
  k_gemm_wmma<<<dim3(HIDDEN/64, BATCH/16), 32, 0, stream>>>(
      zh, LATENT, ew1t /*K rows 0..255*/, HIDDEN, eb1, Pz, 0, nullptr, 0,
      BATCH, HIDDEN, LATENT);
  k_gemm_wmma<<<dim3(HIDDEN/64, (BATCH*MAX_NODES)/16), 32, 0, stream>>>(
      nfh, NODE_F, ew1t + 256 /*K rows 256..383*/, HIDDEN, nullptr, Pi, 0, nullptr, 0,
      BATCH*MAX_NODES, HIDDEN, NODE_F);
  k_gemm_wmma<<<dim3(HIDDEN/64, (BATCH*MAX_NODES)/16), 32, 0, stream>>>(
      nfh, NODE_F, ew1t + 384 /*K rows 384..511*/, HIDDEN, nullptr, Pj, 0, nullptr, 0,
      BATCH*MAX_NODES, HIDDEN, NODE_F);

  /* 4. fused edge MLP (4 waves/block, shared 64-edge LDS tile) */
  k_edge<<<dim3(BATCH, (NEDGE + EDGE_BLK - 1)/EDGE_BLK), 128, 0, stream>>>(
      Pz, Pi, Pj, ew2t, eb2, ew3, eb3, iiW, jjW, out_edges);
}
